// NanochatMiniLM_34368328303158
// MI455X (gfx1250) — matrix operand order repaired and checked by reading, not compiled, not run on hardware
//
#include <hip/hip_runtime.h>
#include <math.h>

#define DEV __device__ __forceinline__

typedef __attribute__((ext_vector_type(16))) _Float16 v16h;
typedef __attribute__((ext_vector_type(8)))  float    v8f;

namespace cfg {
constexpr int B = 2, T = 1024, D = 1024;
constexpr int H = 16, KV = 4, HD = 64;
constexpr int L = 6, VP = 32000;
constexpr int GATE_CH = 12, SMEAR_CH = 24;
constexpr float EPS = 1.1920929e-07f;
constexpr int M = B * T; // 2048 token rows
// |score| <= ||q||*||k|| = (1.2*8*0.125)*(1.2*8) = 11.52. Scores are computed
// directly in the log2 domain (log2e folded into q), shifted by a constant:
// p = exp2(s2 - SHIFT2), p in [2^-22.4, 2^10.9] -> safe in f16 (denorms kept).
constexpr float LOG2E = 1.44269504f;
constexpr float SHIFT2 = 4.0f * LOG2E;
}

DEV v8f wmma32(v16h a, v16h b, v8f c) {
  return __builtin_amdgcn_wmma_f32_16x16x32_f16(false, a, false, b, (short)0, c,
                                                false, false);
}

// ---------------- per-token RMS-style kernels (256 thr, D=1024 -> 4/thr) ----

__global__ __launch_bounds__(256)
void embed_rms_kernel(const int* __restrict__ ids, const float* __restrict__ wte,
                      float* __restrict__ E) {
  int tok = blockIdx.x, tid = threadIdx.x;
  const float* row = wte + (size_t)ids[tok] * cfg::D;
  float vals[4], ss = 0.f;
  #pragma unroll
  for (int i = 0; i < 4; ++i) { float v = row[tid + i * 256]; vals[i] = v; ss += v * v; }
  #pragma unroll
  for (int o = 16; o; o >>= 1) ss += __shfl_xor(ss, o, 32);
  __shared__ float red[8];
  if ((tid & 31) == 0) red[tid >> 5] = ss;
  __syncthreads();
  float tot = 0.f;
  #pragma unroll
  for (int i = 0; i < 8; ++i) tot += red[i];
  float sc = rsqrtf(tot / cfg::D + cfg::EPS);
  #pragma unroll
  for (int i = 0; i < 4; ++i) E[(size_t)tok * cfg::D + tid + i * 256] = vals[i] * sc;
}

__global__ __launch_bounds__(256)
void smear_kernel(const float* __restrict__ E, const float* __restrict__ smear_w,
                  const float* __restrict__ smear_l, float* __restrict__ x,
                  float* __restrict__ x0) {
  int tok = blockIdx.x, tid = threadIdx.x;
  int t = tok % cfg::T;
  const float* e = E + (size_t)tok * cfg::D;
  float g = 0.f;
  if (t > 0) {
    float dot = 0.f;
    #pragma unroll
    for (int c = 0; c < cfg::SMEAR_CH; ++c) dot += e[c] * smear_w[c];
    g = smear_l[0] / (1.f + __expf(-dot));
  }
  const float* ep = E + (size_t)(tok - 1) * cfg::D;
  #pragma unroll
  for (int i = 0; i < 4; ++i) {
    int d = tid + i * 256;
    float v = e[d] + (t > 0 ? g * ep[d] : 0.f);
    x[(size_t)tok * cfg::D + d] = v;
    x0[(size_t)tok * cfg::D + d] = v;
  }
}

// x = a*x + b*x0 (f32, in place); xnh = f16(rms(x))
__global__ __launch_bounds__(256)
void resid_rms_kernel(float* __restrict__ x, const float* __restrict__ x0,
                      const float* __restrict__ resid_l, const float* __restrict__ x0_l,
                      int l, _Float16* __restrict__ xnh) {
  int tok = blockIdx.x, tid = threadIdx.x;
  float a = resid_l[l], b = x0_l[l];
  float vals[4], ss = 0.f;
  #pragma unroll
  for (int i = 0; i < 4; ++i) {
    size_t idx = (size_t)tok * cfg::D + tid + i * 256;
    float v = a * x[idx] + b * x0[idx];
    vals[i] = v; ss += v * v; x[idx] = v;
  }
  #pragma unroll
  for (int o = 16; o; o >>= 1) ss += __shfl_xor(ss, o, 32);
  __shared__ float red[8];
  if ((tid & 31) == 0) red[tid >> 5] = ss;
  __syncthreads();
  float tot = 0.f;
  #pragma unroll
  for (int i = 0; i < 8; ++i) tot += red[i];
  float sc = rsqrtf(tot / cfg::D + cfg::EPS);
  #pragma unroll
  for (int i = 0; i < 4; ++i)
    xnh[(size_t)tok * cfg::D + tid + i * 256] = (_Float16)(vals[i] * sc);
}

__global__ __launch_bounds__(256)
void rms_only_kernel(const float* __restrict__ x, _Float16* __restrict__ xnh) {
  int tok = blockIdx.x, tid = threadIdx.x;
  float vals[4], ss = 0.f;
  #pragma unroll
  for (int i = 0; i < 4; ++i) {
    float v = x[(size_t)tok * cfg::D + tid + i * 256];
    vals[i] = v; ss += v * v;
  }
  #pragma unroll
  for (int o = 16; o; o >>= 1) ss += __shfl_xor(ss, o, 32);
  __shared__ float red[8];
  if ((tid & 31) == 0) red[tid >> 5] = ss;
  __syncthreads();
  float tot = 0.f;
  #pragma unroll
  for (int i = 0; i < 8; ++i) tot += red[i];
  float sc = rsqrtf(tot / cfg::D + cfg::EPS);
  #pragma unroll
  for (int i = 0; i < 4; ++i)
    xnh[(size_t)tok * cfg::D + tid + i * 256] = (_Float16)(vals[i] * sc);
}

__global__ __launch_bounds__(256)
void cvt_h_kernel(const float* __restrict__ src, _Float16* __restrict__ dst) {
  int i = blockIdx.x * 1024 + threadIdx.x;
  #pragma unroll
  for (int j = 0; j < 4; ++j) dst[i + j * 256] = (_Float16)src[i + j * 256];
}

// ---------------- WMMA GEMM: C[M,N] = A[M,K] @ W[N,K]^T --------------------
// A is pre-converted f16 (zero-cvt A fragments); W converted in-loop.
// One wave computes 16x32 (2 N-tiles share the A fragment); 8 waves/block.

constexpr int EPI_STORE = 0, EPI_ADD = 1, EPI_RELU2H = 2, EPI_SOFTCAP = 3,
              EPI_STOREH = 4;

template <int EPI>
__global__ __launch_bounds__(256)
void gemm_wmma_kernel(const _Float16* __restrict__ A, const float* __restrict__ Wt,
                      void* __restrict__ Cv, int N, int K) {
  int wave = threadIdx.x >> 5, lane = threadIdx.x & 31;
  int l16 = lane & 15, hlf = lane >> 4;
  int nt0 = (blockIdx.x * 8 + wave) * 2;   // first of 2 n-tiles
  int mt = blockIdx.y;
  const _Float16* arow = A + (size_t)(mt * 16 + l16) * K; // A lane row = l16
  const float* w0 = Wt + (size_t)(nt0 * 16 + l16) * K;    // B lane col = l16
  const float* w1 = w0 + (size_t)16 * K;
  int ab = hlf * 8, bb = hlf * 16;
  v8f acc0 = {}, acc1 = {};
  for (int kc = 0; kc < K; kc += 32) {
    __builtin_prefetch(arow + kc + 64, 0, 1);
    __builtin_prefetch(w0 + kc + 64, 0, 1);
    __builtin_prefetch(w1 + kc + 64, 0, 1);
    v16h af, bf0, bf1;
    #pragma unroll
    for (int e = 0; e < 8; ++e) {               // A 16x32 f16 lane layout
      af[e]     = arow[kc + bb + e];            // contiguous 16B
      af[e + 8] = arow[kc + bb + 8 + e];        // contiguous 16B
    }
    #pragma unroll
    for (int e = 0; e < 16; ++e) {              // B 32x16 f16 lane layout
      bf0[e] = (_Float16)w0[kc + bb + e];
      bf1[e] = (_Float16)w1[kc + bb + e];
    }
    acc0 = wmma32(af, bf0, acc0);
    acc1 = wmma32(af, bf1, acc1);
  }
  int row0 = mt * 16 + hlf * 8;
  #pragma unroll
  for (int p = 0; p < 2; ++p) {
    int col = (nt0 + p) * 16 + l16;
    #pragma unroll
    for (int r = 0; r < 8; ++r) {               // C layout: vgpr r -> row r(+8)
      size_t idx = (size_t)(row0 + r) * N + col;
      float v = p ? acc1[r] : acc0[r];
      if constexpr (EPI == EPI_ADD) {
        ((float*)Cv)[idx] += v;
      } else if constexpr (EPI == EPI_RELU2H) {
        float t = fmaxf(v, 0.f);
        ((_Float16*)Cv)[idx] = (_Float16)(t * t);
      } else if constexpr (EPI == EPI_SOFTCAP) {
        ((float*)Cv)[idx] = 15.f * tanhf(v * (1.f / 15.f));
      } else if constexpr (EPI == EPI_STOREH) {
        ((_Float16*)Cv)[idx] = (_Float16)v;
      } else {
        ((float*)Cv)[idx] = v;
      }
    }
  }
}

// ---------------- RoPE + RMS (*1.2), f32 in -> f16 out ----------------------

__global__ __launch_bounds__(32)
void rope_rms_kernel(const float* __restrict__ in, _Float16* __restrict__ outh,
                     int nheads, float post) {
  int idx = blockIdx.x;                // (b*T + t)*nheads + h
  int lane = threadIdx.x;              // 32 rotary frequencies
  int t = (idx / nheads) % cfg::T;
  const float* row = in + (size_t)idx * cfg::HD;
  _Float16* orow = outh + (size_t)idx * cfg::HD;
  float inv = powf(100000.f, -(2.f * lane) / 64.f);
  float f = (float)t * inv;
  float c = cosf(f), s = sinf(f);
  float x1 = row[lane], x2 = row[lane + 32];
  float o1 = x1 * c + x2 * s;
  float o2 = -x1 * s + x2 * c;
  float ss = o1 * o1 + o2 * o2;
  #pragma unroll
  for (int o = 16; o; o >>= 1) ss += __shfl_xor(ss, o, 32);
  float sc = 1.2f * post * rsqrtf(ss / 64.f + cfg::EPS);
  orow[lane] = (_Float16)(o1 * sc);
  orow[lane + 32] = (_Float16)(o2 * sc);
}

// ---------------- value-embedding gate add ----------------------------------

__global__ __launch_bounds__(256)
void ve_add_kernel(float* __restrict__ v, const int* __restrict__ ids,
                   const float* __restrict__ ve_tab, const float* __restrict__ ve_gw,
                   const _Float16* __restrict__ xnh) {
  int tok = blockIdx.x, tid = threadIdx.x;    // tid = kv*64 + d
  int kv = tid >> 6;
  const _Float16* xr = xnh + (size_t)tok * cfg::D;
  float dot = 0.f;
  #pragma unroll
  for (int c = 0; c < cfg::GATE_CH; ++c)
    dot += (float)xr[c] * ve_gw[kv * cfg::GATE_CH + c];
  float gate = 3.f / (1.f + __expf(-dot));
  float ve = ve_tab[(size_t)ids[tok] * (cfg::KV * cfg::HD) + tid];
  v[(size_t)tok * 256 + tid] += gate * ve;
}

// ---------------- WMMA attention, windowed causal ---------------------------
// Scores computed in log2 domain (log2e folded into q). Fixed-shift softmax:
// p = exp2(s2 - SHIFT2) -> one v_exp per element, no max, no rescale. Row
// sums via an extra WMMA against a ones matrix (same C-layout rows as O).

__global__ __launch_bounds__(32)
void attn_kernel(const _Float16* __restrict__ q, const _Float16* __restrict__ k,
                 const _Float16* __restrict__ v, _Float16* __restrict__ y, int Wwin) {
  __shared__ _Float16 plds[16 * 32];   // P tile (16q x 32k), C->A repack
  int qt = blockIdx.x, h = blockIdx.y, b = blockIdx.z;
  int lane = threadIdx.x, l16 = lane & 15, hlf = lane >> 4;
  int kvh = h >> 2;                    // GQA: H/KV = 4
  int q0 = qt * 16;
  int ab = hlf * 8, bb = hlf * 16;

  // Q fragments: two K-chunks over HD=64, straight f16 copies
  const _Float16* qrow = q + (((size_t)b * cfg::T + q0 + l16) * cfg::H + h) * cfg::HD;
  v16h qa0, qa1;
  #pragma unroll
  for (int e = 0; e < 8; ++e) {
    qa0[e]     = qrow[bb + e];
    qa0[e + 8] = qrow[bb + 8 + e];
    qa1[e]     = qrow[32 + bb + e];
    qa1[e + 8] = qrow[32 + bb + 8 + e];
  }

  v16h ones;
  #pragma unroll
  for (int e = 0; e < 16; ++e) ones[e] = (_Float16)1.0f;

  v8f oacc[4] = {{}, {}, {}, {}};
  v8f lacc = {};                       // row sums of P (replicated over cols)

  int kstart = q0 - Wwin + 1; if (kstart < 0) kstart = 0; kstart &= ~31;
  int kend = q0 + 15;
  for (int kc = kstart; kc <= kend; kc += 32) {
    // ---- scores S (16q x 32k) via 4 WMMAs ----
    v8f s0 = {}, s1 = {};
    const _Float16* k0p = k + (((size_t)b * cfg::T + kc + l16) * cfg::KV + kvh) * cfg::HD;
    const _Float16* k1p = k0p + 16 * cfg::KV * cfg::HD;
    v16h kf;
    #pragma unroll
    for (int e = 0; e < 16; ++e) kf[e] = k0p[bb + e];
    s0 = wmma32(qa0, kf, s0);
    #pragma unroll
    for (int e = 0; e < 16; ++e) kf[e] = k0p[32 + bb + e];
    s0 = wmma32(qa1, kf, s0);
    #pragma unroll
    for (int e = 0; e < 16; ++e) kf[e] = k1p[bb + e];
    s1 = wmma32(qa0, kf, s1);
    #pragma unroll
    for (int e = 0; e < 16; ++e) kf[e] = k1p[32 + bb + e];
    s1 = wmma32(qa1, kf, s1);

    // ---- masked fixed-shift exp2, store P to LDS ----
    #pragma unroll
    for (int r = 0; r < 8; ++r) {
      int qi = q0 + r + hlf * 8;
      int kk0 = kc + l16, kk1 = kk0 + 16;
      bool ok0 = (kk0 <= qi) && (qi - kk0 < Wwin);
      bool ok1 = (kk1 <= qi) && (qi - kk1 < Wwin);
      float p0 = ok0 ? __builtin_amdgcn_exp2f(s0[r] - cfg::SHIFT2) : 0.f;
      float p1 = ok1 ? __builtin_amdgcn_exp2f(s1[r] - cfg::SHIFT2) : 0.f;
      plds[(r + hlf * 8) * 32 + l16]      = (_Float16)p0;
      plds[(r + hlf * 8) * 32 + l16 + 16] = (_Float16)p1;
    }
    __syncthreads();

    // ---- P (A-frag from LDS): row sums + P x V via 5 WMMAs ----
    v16h pa;
    #pragma unroll
    for (int e = 0; e < 8; ++e) {
      pa[e]     = plds[l16 * 32 + bb + e];
      pa[e + 8] = plds[l16 * 32 + bb + 8 + e];
    }
    lacc = wmma32(pa, ones, lacc);
    const _Float16* vp = v + (((size_t)b * cfg::T + kc + bb) * cfg::KV + kvh) * cfg::HD + l16;
    #pragma unroll
    for (int j = 0; j < 4; ++j) {
      v16h vf;
      #pragma unroll
      for (int e = 0; e < 16; ++e)
        vf[e] = vp[(size_t)e * (cfg::KV * cfg::HD) + j * 16];
      oacc[j] = wmma32(pa, vf, oacc[j]);
    }
    __syncthreads();
  }

  // ---- normalize + store y (B,T,H,HD) as f16 ----
  #pragma unroll
  for (int r = 0; r < 8; ++r) {
    int qi = q0 + r + hlf * 8;
    float rinv = 1.f / lacc[r];
    _Float16* yr = y + (((size_t)b * cfg::T + qi) * cfg::H + h) * cfg::HD + l16;
    #pragma unroll
    for (int j = 0; j < 4; ++j) yr[j * 16] = (_Float16)(oacc[j][r] * rinv);
  }
}

// ---------------- host-side sequencing --------------------------------------

extern "C" void kernel_launch(void* const* d_in, const int* in_sizes, int n_in,
                              void* d_out, int out_size, void* d_ws, size_t ws_size,
                              hipStream_t stream) {
  using namespace cfg;
  (void)in_sizes; (void)n_in; (void)out_size; (void)ws_size;
  const int*   ids     = (const int*)d_in[0];
  const float* wte     = (const float*)d_in[1];
  const float* Wq      = (const float*)d_in[2];
  const float* Wk      = (const float*)d_in[3];
  const float* Wv      = (const float*)d_in[4];
  const float* Wo      = (const float*)d_in[5];
  const float* Wfc     = (const float*)d_in[6];
  const float* Wmp     = (const float*)d_in[7];
  const float* ve_tab  = (const float*)d_in[8];
  const float* ve_gw   = (const float*)d_in[9];
  const float* lm_w    = (const float*)d_in[10];
  const float* resid_l = (const float*)d_in[11];
  const float* x0_l    = (const float*)d_in[12];
  const float* smear_w = (const float*)d_in[13];
  const float* smear_l = (const float*)d_in[14];
  float* out = (float*)d_out;

  char* ws = (char*)d_ws;
  size_t off = 0;
  auto allocf = [&](size_t n) { float* p = (float*)(ws + off); off += n * 4; return p; };
  auto alloch = [&](size_t n) { _Float16* p = (_Float16*)(ws + off); off += n * 2;
                                off = (off + 255) & ~(size_t)255; return p; };
  float*     x    = allocf((size_t)M * D);
  float*     x0   = allocf((size_t)M * D);
  float*     qb   = allocf((size_t)M * H * HD);
  float*     kb   = allocf((size_t)M * KV * HD);
  float*     vb   = allocf((size_t)M * KV * HD);
  float*     Eb   = allocf((size_t)M * D);
  _Float16*  xnh  = alloch((size_t)M * D);
  _Float16*  qh   = alloch((size_t)M * H * HD);
  _Float16*  kh   = alloch((size_t)M * KV * HD);
  _Float16*  vh   = alloch((size_t)M * KV * HD);
  _Float16*  ybh  = alloch((size_t)M * H * HD);
  _Float16*  hidh = alloch((size_t)M * 4 * D);

  const int WIN[L] = {256, 1024, 256, 1024, 256, 1024};
  const int VEI[L] = {-1, 0, -1, 1, -1, 2};

  embed_rms_kernel<<<M, 256, 0, stream>>>(ids, wte, Eb);
  smear_kernel<<<M, 256, 0, stream>>>(Eb, smear_w, smear_l, x, x0);

  for (int l = 0; l < L; ++l) {
    resid_rms_kernel<<<M, 256, 0, stream>>>(x, x0, resid_l, x0_l, l, xnh);
    gemm_wmma_kernel<EPI_STORE><<<dim3((H * HD) / 256, M / 16), 256, 0, stream>>>(
        xnh, Wq + (size_t)l * H * HD * D, qb, H * HD, D);
    gemm_wmma_kernel<EPI_STORE><<<dim3((KV * HD) / 256, M / 16), 256, 0, stream>>>(
        xnh, Wk + (size_t)l * KV * HD * D, kb, KV * HD, D);
    if (VEI[l] >= 0) {
      // V in f32, add gated value-embeddings, then convert to f16
      gemm_wmma_kernel<EPI_STORE><<<dim3((KV * HD) / 256, M / 16), 256, 0, stream>>>(
          xnh, Wv + (size_t)l * KV * HD * D, vb, KV * HD, D);
      ve_add_kernel<<<M, 256, 0, stream>>>(
          vb, ids, ve_tab + (size_t)VEI[l] * VP * KV * HD,
          ve_gw + (size_t)VEI[l] * KV * GATE_CH, xnh);
      cvt_h_kernel<<<(M * KV * HD) / 1024, 256, 0, stream>>>(vb, vh);
    } else {
      // no value-embeddings: V straight to f16
      gemm_wmma_kernel<EPI_STOREH><<<dim3((KV * HD) / 256, M / 16), 256, 0, stream>>>(
          xnh, Wv + (size_t)l * KV * HD * D, vh, KV * HD, D);
    }
    // q carries 1/sqrt(HD) and log2e (scores land in log2 domain)
    rope_rms_kernel<<<M * H, 32, 0, stream>>>(qb, qh, H, 0.125f * LOG2E);
    rope_rms_kernel<<<M * KV, 32, 0, stream>>>(kb, kh, KV, 1.0f);
    attn_kernel<<<dim3(T / 16, H, B), 32, 0, stream>>>(qh, kh, vh, ybh, WIN[l]);
    gemm_wmma_kernel<EPI_ADD><<<dim3(D / 256, M / 16), 256, 0, stream>>>(
        ybh, Wo + (size_t)l * D * D, x, D, D);
    rms_only_kernel<<<M, 256, 0, stream>>>(x, xnh);
    gemm_wmma_kernel<EPI_RELU2H><<<dim3((4 * D) / 256, M / 16), 256, 0, stream>>>(
        xnh, Wfc + (size_t)l * 4 * D * D, hidh, 4 * D, D);
    gemm_wmma_kernel<EPI_ADD><<<dim3(D / 256, M / 16), 256, 0, stream>>>(
        hidh, Wmp + (size_t)l * D * 4 * D, x, D, 4 * D);
  }

  rms_only_kernel<<<M, 256, 0, stream>>>(x, xnh);
  gemm_wmma_kernel<EPI_SOFTCAP><<<dim3(VP / 256, M / 16), 256, 0, stream>>>(
      xnh, lm_w, out, VP, D);
}
